// MultimodalAttentionDecoder_54314156425581
// MI455X (gfx1250) — compile-verified, matrix operand order checked by reading, not run on hardware
//
#include <hip/hip_runtime.h>
#include <hip/hip_bf16.h>
#include <math.h>

// MI455X / gfx1250: wave32, WMMA 16x16x32 bf16 with f32 accum.

typedef __attribute__((ext_vector_type(16))) __bf16 bf16x16;
typedef __attribute__((ext_vector_type(8)))  float  f32x8;

#define S_LEN 1024
#define N2H   2048

static __device__ __forceinline__ float sigmoidf_(float x) {
  return 1.0f / (1.0f + __expf(-x));
}

// Load 16 consecutive f32 into 4 float4 registers.
static __device__ __forceinline__ void ld16(const float* __restrict__ src, float4* r) {
  const float4* s4 = reinterpret_cast<const float4*>(src);
  r[0] = s4[0]; r[1] = s4[1]; r[2] = s4[2]; r[3] = s4[3];
}

// Convert 4 float4 regs -> 16 bf16, store contiguously to LDS (16B aligned).
static __device__ __forceinline__ void st16_bf16(const float4* r, __bf16* dst) {
  union { __bf16 h[16]; uint4 q[2]; } u;
#pragma unroll
  for (int i = 0; i < 4; ++i) {
    float4 f = r[i];
    u.h[4 * i + 0] = (__bf16)f.x;
    u.h[4 * i + 1] = (__bf16)f.y;
    u.h[4 * i + 2] = (__bf16)f.z;
    u.h[4 * i + 3] = (__bf16)f.w;
  }
  reinterpret_cast<uint4*>(dst)[0] = u.q[0];
  reinterpret_cast<uint4*>(dst)[1] = u.q[1];
}

// Load a 16x32 bf16 WMMA operand fragment from LDS.
// p points at row start + koff; ISA layout: elems 0..7 = K[koff..koff+7],
// elems 8..15 = K[koff+16..koff+23].
static __device__ __forceinline__ bf16x16 frag_ld(const __bf16* p) {
  union { bf16x16 v; uint4 q[2]; } u;
  u.q[0] = *reinterpret_cast<const uint4*>(p);
  u.q[1] = *reinterpret_cast<const uint4*>(p + 16);
  return u.v;
}

// ---------------------------------------------------------------------------
// Heavy fused kernel:  e[b,s] (+)= sum_c v_w[c] * tanh( (X @ W^T)[bs,c]
//                         + biasbh[b,c] + wlin_b[c] + wc_b[c] + wc_w[c]*cov[bs] )
// X: (B*S, 2048) f32 row-major, W: (2048, 2048) f32 row-major (C = X W^T).
// Tile 128x128, BK=32, 8 waves, each wave 64x32 (4x2 WMMA tiles).
// Software-pipelined: next tile's global loads issued before the WMMA block.
// ---------------------------------------------------------------------------
__global__ __launch_bounds__(256) void attn_score_gemm(
    const float* __restrict__ X, const float* __restrict__ W,
    const float* __restrict__ biasbh, const float* __restrict__ wlin_b,
    const float* __restrict__ wc_b, const float* __restrict__ wc_w,
    const float* __restrict__ v_w, const float* __restrict__ cov,
    float* __restrict__ e_out) {
  constexpr int BM = 128, BN = 128, BK = 32, PITCH = 40;  // pitch pad: no bank conflicts
  __shared__ __align__(16) __bf16 As[BM * PITCH];
  __shared__ __align__(16) __bf16 Bs[BN * PITCH];
  __shared__ float s_cov[BM], s_bias[BN], s_wc[BN], s_vw[BN], e_part[BM];

  const int tid = threadIdx.x;
  const int lane = tid & 31, wave = tid >> 5;
  const int wm = wave >> 2, wn = wave & 3;           // 2 x 4 wave grid
  const int row_base = blockIdx.y * BM;
  const int col_base = blockIdx.x * BN;
  const int b = row_base >> 10;                      // S=1024 divides tile rows

  if (tid < BM) {
    s_cov[tid] = cov[row_base + tid];
    e_part[tid] = 0.0f;
  } else {
    int t = tid - BM;
    int c = col_base + t;
    s_bias[t] = biasbh[b * N2H + c] + wlin_b[c] + wc_b[c];
    s_wc[t] = wc_w[c];
    s_vw[t] = v_w[c];
  }

  f32x8 acc[4][2];
  const f32x8 zf = {};
#pragma unroll
  for (int mt = 0; mt < 4; ++mt)
#pragma unroll
    for (int nt = 0; nt < 2; ++nt) acc[mt][nt] = zf;

  const int lr = tid >> 1;          // 0..127 : tile row this thread loads
  const int lks = (tid & 1) * 16;   // 0 / 16 : k-half this thread loads
  const int frow = lane & 15;
  const int koff = (lane >> 4) * 8;

  const float* pA = X + (size_t)(row_base + lr) * N2H + lks;
  const float* pB = W + (size_t)(col_base + lr) * N2H + lks;
  __bf16* sA = &As[lr * PITCH + lks];
  __bf16* sB = &Bs[lr * PITCH + lks];

  float4 ax[4], bx[4];
  ld16(pA, ax);                     // prologue: tile for kk=0
  ld16(pB, bx);

  for (int kk = 0; kk < N2H; kk += BK) {
    __syncthreads();                // previous tile fully consumed
    st16_bf16(ax, sA);              // stage current tile (regs -> bf16 LDS)
    st16_bf16(bx, sB);
    __syncthreads();                // tile visible to all waves

    bf16x16 af[4], bfr[2];
#pragma unroll
    for (int mt = 0; mt < 4; ++mt)
      af[mt] = frag_ld(&As[(wm * 64 + mt * 16 + frow) * PITCH + koff]);
#pragma unroll
    for (int nt = 0; nt < 2; ++nt)
      bfr[nt] = frag_ld(&Bs[(wn * 32 + nt * 16 + frow) * PITCH + koff]);

    if (kk + BK < N2H) {            // prefetch next tile; latency hidden by WMMAs
      ld16(pA + kk + BK, ax);
      ld16(pB + kk + BK, bx);
    }

#pragma unroll
    for (int mt = 0; mt < 4; ++mt)
#pragma unroll
      for (int nt = 0; nt < 2; ++nt)
        acc[mt][nt] = __builtin_amdgcn_wmma_f32_16x16x32_bf16(
            false, af[mt], false, bfr[nt], (short)0, acc[mt][nt], false, false);
  }

  // Fused epilogue: tanh + dot(v_w) + row reduction.
  // C layout: lane<16 -> N=lane, M=i ; lane>=16 -> N=lane-16, M=8+i.
#pragma unroll
  for (int mt = 0; mt < 4; ++mt) {
#pragma unroll
    for (int nt = 0; nt < 2; ++nt) {
      const int cl = wn * 32 + nt * 16 + (lane & 15);
      const float bia = s_bias[cl];
      const float wcv = s_wc[cl];
      const float vwv = s_vw[cl];
#pragma unroll
      for (int i = 0; i < 8; ++i) {
        int rl = wm * 64 + mt * 16 + ((lane >> 4) << 3) + i;
        float val = acc[mt][nt][i] + bia + wcv * s_cov[rl];
        float contrib = tanhf(val) * vwv;
#pragma unroll
        for (int m = 1; m < 16; m <<= 1) contrib += __shfl_xor(contrib, m, 16);
        if ((lane & 15) == 0) atomicAdd(&e_part[rl], contrib);
      }
    }
  }
  __syncthreads();
  if (tid < BM) atomicAdd(&e_out[row_base + tid], e_part[tid]);
}

// ---------------------------------------------------------------------------
// Skinny GEMM with K-split: Y(16 x N) += X(16 x K) @ W(N x K)^T [+ bias]
// grid.x tiles N by 64 (4 waves x 16 cols); grid.y splits K; partial results
// atomically accumulated into zero-initialized Y. bias added by y==0 slice.
// ---------------------------------------------------------------------------
__global__ __launch_bounds__(128) void gemm16(
    const float* __restrict__ X, const float* __restrict__ W,
    const float* __restrict__ bias, float* __restrict__ Y,
    int N, int K) {
  constexpr int PITCH = 40;
  __shared__ __align__(16) __bf16 As[16 * PITCH];

  const int tid = threadIdx.x;
  const int lane = tid & 31, wave = tid >> 5;
  const int n = blockIdx.x * 64 + wave * 16 + (lane & 15);
  const int koff = (lane >> 4) * 8;

  const int chunk = K / gridDim.y;
  const int k0 = blockIdx.y * chunk;
  const int k1 = k0 + chunk;

  f32x8 acc = {};
  const int lr = tid >> 3;          // 0..15
  const int lks = (tid & 7) * 4;    // 0..28 step 4

  const float* pA = X + (size_t)lr * K + lks;
  const float* pW = W + (size_t)n * K + koff;
  __bf16* sA = &As[lr * PITCH + lks];

  float4 aR;
  float4 bR[4];
  {  // prologue loads for k0
    aR = *reinterpret_cast<const float4*>(pA + k0);
    const float* wp = pW + k0;
    bR[0] = *reinterpret_cast<const float4*>(wp);
    bR[1] = *reinterpret_cast<const float4*>(wp + 4);
    bR[2] = *reinterpret_cast<const float4*>(wp + 16);
    bR[3] = *reinterpret_cast<const float4*>(wp + 20);
  }

  for (int kk = k0; kk < k1; kk += 32) {
    __syncthreads();
    {
      union { __bf16 h[4]; uint2 q; } u;
      u.h[0] = (__bf16)aR.x; u.h[1] = (__bf16)aR.y;
      u.h[2] = (__bf16)aR.z; u.h[3] = (__bf16)aR.w;
      *reinterpret_cast<uint2*>(sA) = u.q;
    }
    __syncthreads();

    bf16x16 af = frag_ld(&As[(lane & 15) * PITCH + koff]);

    union { bf16x16 v; __bf16 h[16]; } ub;
    ub.h[0] = (__bf16)bR[0].x;  ub.h[1] = (__bf16)bR[0].y;
    ub.h[2] = (__bf16)bR[0].z;  ub.h[3] = (__bf16)bR[0].w;
    ub.h[4] = (__bf16)bR[1].x;  ub.h[5] = (__bf16)bR[1].y;
    ub.h[6] = (__bf16)bR[1].z;  ub.h[7] = (__bf16)bR[1].w;
    ub.h[8] = (__bf16)bR[2].x;  ub.h[9] = (__bf16)bR[2].y;
    ub.h[10] = (__bf16)bR[2].z; ub.h[11] = (__bf16)bR[2].w;
    ub.h[12] = (__bf16)bR[3].x; ub.h[13] = (__bf16)bR[3].y;
    ub.h[14] = (__bf16)bR[3].z; ub.h[15] = (__bf16)bR[3].w;

    if (kk + 32 < k1) {  // prefetch next chunk
      aR = *reinterpret_cast<const float4*>(pA + kk + 32);
      const float* wp = pW + kk + 32;
      bR[0] = *reinterpret_cast<const float4*>(wp);
      bR[1] = *reinterpret_cast<const float4*>(wp + 4);
      bR[2] = *reinterpret_cast<const float4*>(wp + 16);
      bR[3] = *reinterpret_cast<const float4*>(wp + 20);
    }

    acc = __builtin_amdgcn_wmma_f32_16x16x32_bf16(false, af, false, ub.v,
                                                  (short)0, acc, false, false);
  }

  const float bv = (blockIdx.y == 0 && bias) ? bias[n] : 0.0f;
#pragma unroll
  for (int i = 0; i < 8; ++i) {
    int m = ((lane >> 4) << 3) + i;
    atomicAdd(&Y[(size_t)m * N + n], acc[i] + bv);
  }
}

// ---------------------------------------------------------------------------
// Small support kernels
// ---------------------------------------------------------------------------
__global__ void init_kernel(float* e1, float* e2, float* c1, float* c2,
                            float* bias1, float* bias2, float* z1, float* z2,
                            float* gates, float* logits,
                            const float* v1b, const float* v2b) {
  int i = blockIdx.x * 256 + threadIdx.x;  // grid covers 65536
  gates[i] = 0.0f;
  if (i < 32768) {
    c1[i] = 0.0f; c2[i] = 0.0f;
    bias1[i] = 0.0f; bias2[i] = 0.0f;
    z1[i] = 0.0f; z2[i] = 0.0f;
    logits[i] = 0.0f;
  }
  if (i < 16384) { e1[i] = v1b[0]; e2[i] = v2b[0]; }
}

__global__ void softmax_rows(const float* __restrict__ in, float* __restrict__ out,
                             int C) {
  __shared__ float red[256];
  const int row = blockIdx.x, tid = threadIdx.x;
  const float* p = in + (size_t)row * C;
  float m = -1e30f;
  for (int c = tid; c < C; c += 256) m = fmaxf(m, p[c]);
  red[tid] = m; __syncthreads();
  for (int s = 128; s > 0; s >>= 1) {
    if (tid < s) red[tid] = fmaxf(red[tid], red[tid + s]);
    __syncthreads();
  }
  m = red[0]; __syncthreads();
  float sum = 0.0f;
  for (int c = tid; c < C; c += 256) sum += __expf(p[c] - m);
  red[tid] = sum; __syncthreads();
  for (int s = 128; s > 0; s >>= 1) {
    if (tid < s) red[tid] += red[tid + s];
    __syncthreads();
  }
  const float inv = 1.0f / red[0];
  for (int c = tid; c < C; c += 256) out[(size_t)row * C + c] = __expf(p[c] - m) * inv;
}

// c[b,:] += sum_s a[b,s] * X[b,s,:]  (partial over s-chunk, atomics into c)
__global__ void context_kernel(const float* __restrict__ a,
                               const float* __restrict__ X,
                               float* __restrict__ cacc) {
  __shared__ float sa[256];
  const int b = blockIdx.z, sc = blockIdx.y, jc = blockIdx.x, tid = threadIdx.x;
  sa[tid] = a[b * S_LEN + sc * 256 + tid];
  __syncthreads();
  const int j = jc * 1024 + tid * 4;
  float4 acc = {0.f, 0.f, 0.f, 0.f};
  const float* base = X + ((size_t)b * S_LEN + sc * 256) * N2H + j;
  for (int i = 0; i < 256; ++i) {
    float4 v = *reinterpret_cast<const float4*>(base + (size_t)i * N2H);
    float w = sa[i];
    acc.x += w * v.x; acc.y += w * v.y; acc.z += w * v.z; acc.w += w * v.w;
  }
  atomicAdd(&cacc[b * N2H + j + 0], acc.x);
  atomicAdd(&cacc[b * N2H + j + 1], acc.y);
  atomicAdd(&cacc[b * N2H + j + 2], acc.z);
  atomicAdd(&cacc[b * N2H + j + 3], acc.w);
}

__global__ void csum_kernel(const float* __restrict__ c, float* __restrict__ cs) {
  int j = blockIdx.x * 256 + threadIdx.x;
  float s = 0.0f;
  for (int b = 0; b < 16; ++b) s += c[b * N2H + j];
  cs[j] = s;
}

// eb[b] = vb_b + sum_j vb_w[j] * tanh(z[b,j])
__global__ void ebreduce_kernel(const float* __restrict__ z,
                                const float* __restrict__ vbw,
                                const float* __restrict__ vbb,
                                float* __restrict__ out) {
  __shared__ float red[256];
  const int b = blockIdx.x, tid = threadIdx.x;
  float s = 0.0f;
  for (int j = tid; j < N2H; j += 256) s += vbw[j] * tanhf(z[b * N2H + j]);
  red[tid] = s; __syncthreads();
  for (int k = 128; k > 0; k >>= 1) {
    if (tid < k) red[tid] += red[tid + k];
    __syncthreads();
  }
  if (tid == 0) out[b] = red[0] + vbb[0];
}

__global__ void attcov_kernel(const float* __restrict__ eb1, const float* __restrict__ eb2,
                              const float* __restrict__ a1, const float* __restrict__ a2,
                              const float* __restrict__ cov,
                              float* __restrict__ out_att, float* __restrict__ out_cov) {
  int i = blockIdx.x * 256 + threadIdx.x;  // 16384
  int b = i >> 10;
  float att = eb1[b] * a1[i] + eb2[b] * a2[i];
  out_att[i] = att;
  out_cov[i] = cov[i] + att;
}

// c3 = eb1[b]*csum1[j]*c1 + eb2[b]*csum2[j]*c2 ; x = [c3 | sent_embed]
__global__ void combine_kernel(const float* __restrict__ eb1, const float* __restrict__ eb2,
                               const float* __restrict__ cs1, const float* __restrict__ cs2,
                               const float* __restrict__ c1, const float* __restrict__ c2,
                               const float* __restrict__ sent, float* __restrict__ x) {
  int i = blockIdx.x * 256 + threadIdx.x;  // 32768
  int b = i >> 11, j = i & 2047;
  float c3 = eb1[b] * cs1[j] * c1[i] + eb2[b] * cs2[j] * c2[i];
  x[(size_t)b * 4096 + j] = c3;
  x[(size_t)b * 4096 + 2048 + j] = sent[i];
}

__global__ void lstm_kernel(const float* __restrict__ gates, const float* __restrict__ c0,
                            float* __restrict__ hnew, float* __restrict__ cnew) {
  int i = blockIdx.x * 256 + threadIdx.x;  // 16384
  int b = i >> 10, k = i & 1023;
  const float* g = gates + (size_t)b * 4096;
  float ig = sigmoidf_(g[k]);
  float fg = sigmoidf_(g[1024 + k]);
  float gg = tanhf(g[2048 + k]);
  float og = sigmoidf_(g[3072 + k]);
  float cn = fg * c0[i] + ig * gg;
  float hn = og * tanhf(cn);
  cnew[i] = cn;
  hnew[i] = hn;
}

// ---------------------------------------------------------------------------
extern "C" void kernel_launch(void* const* d_in, const int* in_sizes, int n_in,
                              void* d_out, int out_size, void* d_ws, size_t ws_size,
                              hipStream_t stream) {
  (void)in_sizes; (void)n_in; (void)out_size; (void)ws_size;

  const float* sent  = (const float*)d_in[0];
  const float* h0    = (const float*)d_in[1];   // (16,1024)
  const float* c0    = (const float*)d_in[2];   // (16,1024)
  const float* Xa    = (const float*)d_in[3];   // (16,1024,2048)
  const float* Xi    = (const float*)d_in[4];
  const float* cov   = (const float*)d_in[5];   // (16,1024)
  const float *W1w = (const float*)d_in[6],  *W1b = (const float*)d_in[7];
  const float *W3w = (const float*)d_in[8],  *W3b = (const float*)d_in[9];
  const float *Wb1w = (const float*)d_in[10], *Wb1b = (const float*)d_in[11];
  const float *Wb3w = (const float*)d_in[12], *Wb3b = (const float*)d_in[13];
  const float *W2w = (const float*)d_in[14], *W2b = (const float*)d_in[15];
  const float *W4w = (const float*)d_in[16], *W4b = (const float*)d_in[17];
  const float *Wb2w = (const float*)d_in[18], *Wb2b = (const float*)d_in[19];
  const float *Wb4w = (const float*)d_in[20], *Wb4b = (const float*)d_in[21];
  const float *Wc1w = (const float*)d_in[22], *Wc1b = (const float*)d_in[23];
  const float *Wc2w = (const float*)d_in[24], *Wc2b = (const float*)d_in[25];
  const float *v1w = (const float*)d_in[26], *v1b = (const float*)d_in[27];
  const float *v2w = (const float*)d_in[28], *v2b = (const float*)d_in[29];
  const float *vb1w = (const float*)d_in[30], *vb1b = (const float*)d_in[31];
  const float *vb2w = (const float*)d_in[32], *vb2b = (const float*)d_in[33];
  const float *Wihw = (const float*)d_in[34], *Wihb = (const float*)d_in[35];
  const float *Whhw = (const float*)d_in[36], *Whhb = (const float*)d_in[37];
  const float *Woutw = (const float*)d_in[38], *Woutb = (const float*)d_in[39];

  float* ws = (float*)d_ws;
  float* bias1 = ws + 0;        // 16x2048 (atomic acc)
  float* bias2 = ws + 32768;    // 16x2048 (atomic acc)
  float* e1    = ws + 65536;    // 16x1024 (atomic acc, seeded v1_b)
  float* e2    = ws + 81920;
  float* a1    = ws + 98304;    // 16x1024
  float* a2    = ws + 114688;
  float* c1    = ws + 131072;   // 16x2048 (atomic acc)
  float* c2    = ws + 163840;
  float* cs1   = ws + 196608;   // 2048
  float* cs2   = ws + 198656;
  float* z1    = ws + 200704;   // 16x2048 (atomic acc)
  float* z2    = ws + 233472;
  float* eb1   = ws + 266240;   // 16
  float* eb2   = ws + 266256;   // 16
  float* x     = ws + 266304;   // 16x4096
  float* gates = ws + 331840;   // 16x4096 (atomic acc)
  float* logits = ws + 397376;  // 16x2048 (atomic acc)

  float* out_final = (float*)d_out;        // (16,2048) softmax
  float* out_h   = out_final + 32768;      // (16,1,1024)
  float* out_c   = out_final + 49152;      // (1,16,1024)
  float* out_att = out_final + 65536;      // (16,1024,1)
  float* out_cov = out_final + 81920;      // (16,1024,1)

  dim3 blk(256);

  init_kernel<<<256, blk, 0, stream>>>(e1, e2, c1, c2, bias1, bias2, z1, z2,
                                       gates, logits, v1b, v2b);

  // Per-(b,col) bias vectors: bias1 = h0 @ W2^T + W2_b ; bias2 = h0 @ W4^T + W4_b
  gemm16<<<dim3(32, 2), 128, 0, stream>>>(h0, W2w, W2b, bias1, 2048, 1024);
  gemm16<<<dim3(32, 2), 128, 0, stream>>>(h0, W4w, W4b, bias2, 2048, 1024);

  // Fused attention score GEMMs (the 2x137 GFLOP hot spot)
  attn_score_gemm<<<dim3(16, 128), blk, 0, stream>>>(Xa, W1w, bias1, W1b, Wc1b,
                                                     Wc1w, v1w, cov, e1);
  attn_score_gemm<<<dim3(16, 128), blk, 0, stream>>>(Xi, W3w, bias2, W3b, Wc2b,
                                                     Wc2w, v2w, cov, e2);

  softmax_rows<<<16, blk, 0, stream>>>(e1, a1, 1024);
  softmax_rows<<<16, blk, 0, stream>>>(e2, a2, 1024);

  context_kernel<<<dim3(2, 4, 16), blk, 0, stream>>>(a1, Xa, c1);
  context_kernel<<<dim3(2, 4, 16), blk, 0, stream>>>(a2, Xi, c2);

  csum_kernel<<<8, blk, 0, stream>>>(c1, cs1);
  csum_kernel<<<8, blk, 0, stream>>>(c2, cs2);

  // z1 = c1@Wb1^T + Wb1_b + h0@Wb2^T + Wb2_b ; z2 analog (atomic accumulate)
  gemm16<<<dim3(32, 4), 128, 0, stream>>>(c1, Wb1w, Wb1b, z1, 2048, 2048);
  gemm16<<<dim3(32, 2), 128, 0, stream>>>(h0, Wb2w, Wb2b, z1, 2048, 1024);
  gemm16<<<dim3(32, 4), 128, 0, stream>>>(c2, Wb3w, Wb3b, z2, 2048, 2048);
  gemm16<<<dim3(32, 2), 128, 0, stream>>>(h0, Wb4w, Wb4b, z2, 2048, 1024);

  ebreduce_kernel<<<16, blk, 0, stream>>>(z1, vb1w, vb1b, eb1);
  ebreduce_kernel<<<16, blk, 0, stream>>>(z2, vb2w, vb2b, eb2);

  attcov_kernel<<<64, blk, 0, stream>>>(eb1, eb2, a1, a2, cov, out_att, out_cov);
  combine_kernel<<<128, blk, 0, stream>>>(eb1, eb2, cs1, cs2, c1, c2, sent, x);

  // LSTM gates = x@Wih^T + Wih_b + h0@Whh^T + Whh_b
  gemm16<<<dim3(64, 4), 128, 0, stream>>>(x, Wihw, Wihb, gates, 4096, 4096);
  gemm16<<<dim3(64, 2), 128, 0, stream>>>(h0, Whhw, Whhb, gates, 4096, 1024);
  lstm_kernel<<<64, blk, 0, stream>>>(gates, c0, out_h, out_c);

  // final_out = softmax(h_new @ Wout^T + Wout_b)
  gemm16<<<dim3(32, 2), 128, 0, stream>>>(out_h, Woutw, Woutb, logits, 2048, 1024);
  softmax_rows<<<16, blk, 0, stream>>>(logits, out_final, 2048);
}
